// SoftQNet_1803886264673
// MI455X (gfx1250) — matrix operand order repaired and checked by reading, not compiled, not run on hardware
//
#include <hip/hip_runtime.h>

typedef _Float16 half8  __attribute__((ext_vector_type(8)));
typedef _Float16 half16 __attribute__((ext_vector_type(16)));
typedef float    float8 __attribute__((ext_vector_type(8)));

#define B_TOT   65536
#define T_STEPS 50
#define LH      32
#define NG      128      // 4*LH gates
#define NS      24
#define NA      6
#define HDIM    256

// gfx1250 has V_TANH_F32 (hardware transcendental, co-executes with WMMA).
__device__ __forceinline__ float fast_tanh(float x) {
#if __has_builtin(__builtin_amdgcn_tanhf)
    return __builtin_amdgcn_tanhf(x);
#else
    return tanhf(x);
#endif
}
// sigmoid(x) = 0.5*tanh(x/2) + 0.5  -> one trans op + mul + fma
__device__ __forceinline__ float fast_sig(float x) {
    return __builtin_fmaf(0.5f, fast_tanh(0.5f * x), 0.5f);
}

__device__ __forceinline__ float8 wmma_f16(half16 a, half16 b, float8 c) {
    // v_wmma_f32_16x16x32_f16: D = A(16x32 f16) * B(32x16 f16) + C(16x16 f32)
    return __builtin_amdgcn_wmma_f32_16x16x32_f16(false, a, false, b, (short)0, c, false, false);
}

// A-matrix fragment, 16x32 f16, row-major source with row stride `stride` halves.
// Layout: lanes 0-15 (M=lane) hold K=0..15;
//         lanes 16-31 (M=lane-16) hold K=16..31.
__device__ __forceinline__ half16 load_fragA(const _Float16* base, int stride, int lane) {
    int m  = lane & 15;
    int k0 = (lane < 16) ? 0 : 16;
    half8 lo = *(const half8*)(base + m * stride + k0);
    half8 hi = *(const half8*)(base + m * stride + k0 + 8);
    return __builtin_shufflevector(lo, hi, 0,1,2,3,4,5,6,7,8,9,10,11,12,13,14,15);
}

// B-matrix fragment, 32x16 f16, B[k][n] = W[n][k], W row-major with row length `krow` halves.
// Layout (contiguous k, same as load_fragA): lane l holds column n=l&15; lanes 0-15 hold K=0..15, lanes 16-31 hold K=16..31.
__device__ __forceinline__ half16 load_fragB(const _Float16* Wrow, int krow, int lane) {
    int n  = lane & 15;
    int k0 = (lane < 16) ? 0 : 16;
    half8 lo = *(const half8*)(Wrow + n * krow + k0);
    half8 hi = *(const half8*)(Wrow + n * krow + k0 + 8);
    return __builtin_shufflevector(lo, hi, 0,1,2,3,4,5,6,7,8,9,10,11,12,13,14,15);
}

// ---------------------------------------------------------------------------
// Kernel 1: fused 2-layer LSTM scan + final 32->2 linear. One wave = 16 rows.
// ---------------------------------------------------------------------------
__global__ __launch_bounds__(256)
void lstm_kernel(const float* __restrict__ xseq,
                 const float* __restrict__ Wih0, const float* __restrict__ Whh0,
                 const float* __restrict__ bih0, const float* __restrict__ bhh0,
                 const float* __restrict__ Wih1, const float* __restrict__ Whh1,
                 const float* __restrict__ bih1, const float* __restrict__ bhh1,
                 const float* __restrict__ Wl,   const float* __restrict__ bl,
                 float* __restrict__ outL)
{
    __shared__ _Float16 sWhh0[NG * LH];
    __shared__ _Float16 sWih1[NG * LH];
    __shared__ _Float16 sWhh1[NG * LH];
    __shared__ float    sX[128 * 2];
    __shared__ _Float16 sH0[8][16 * LH];
    __shared__ _Float16 sH1[8][16 * LH];

    const int tid   = threadIdx.x;
    const int wave  = tid >> 5;
    const int lane  = tid & 31;
    const int gbase = blockIdx.x * 128;
    const int moff  = (lane < 16) ? 0 : 8;

    // Stage weights to LDS as f16; zero-init hidden state tiles.
    for (int i = tid; i < NG * LH; i += 256) {
        sWhh0[i] = (_Float16)Whh0[i];
        sWih1[i] = (_Float16)Wih1[i];
        sWhh1[i] = (_Float16)Whh1[i];
    }
    for (int i = tid; i < 8 * 16 * LH; i += 256) {
        (&sH0[0][0])[i] = (_Float16)0.0f;
        (&sH1[0][0])[i] = (_Float16)0.0f;
    }

    // Per-lane loop-invariant constants: gate n = t8*16 + (lane&15)
    float w0c[8], w1c[8], bs0[8], bs1[8];
#pragma unroll
    for (int t8 = 0; t8 < 8; ++t8) {
        int n   = t8 * 16 + (lane & 15);
        w0c[t8] = Wih0[n * 2 + 0];
        w1c[t8] = Wih0[n * 2 + 1];
        bs0[t8] = bih0[n] + bhh0[n];
        bs1[t8] = bih1[n] + bhh1[n];
    }

    // Cell state in C-layout registers (tile jh covers hidden units jh*16..jh*16+15)
    float8 c0[2], c1[2];
#pragma unroll
    for (int jh = 0; jh < 2; ++jh) { c0[jh] = (float8){}; c1[jh] = (float8){}; }

    for (int t = 0; t < T_STEPS; ++t) {
        __syncthreads();
        {   // cooperatively stage x_t for the block's 128 rows
            int s = tid >> 1, comp = tid & 1;
            sX[s * 2 + comp] = xseq[(size_t)(gbase + s) * (T_STEPS * 2) + t * 2 + comp];
            if (t + 1 < T_STEPS)  // prefetch next step's slice into L2
                __builtin_prefetch(&xseq[(size_t)(gbase + s) * (T_STEPS * 2) + (t + 1) * 2 + comp], 0, 1);
        }
        __syncthreads();

        float x0[8], x1[8];
#pragma unroll
        for (int r = 0; r < 8; ++r) {
            int ml = wave * 16 + moff + r;
            x0[r] = sX[ml * 2 + 0];
            x1[r] = sX[ml * 2 + 1];
        }

        // ---- layer 0: gates = bias + x@Wih0^T (VALU, K=2) + h0@Whh0^T (WMMA, K=32)
        half16 a0 = load_fragA(&sH0[wave][0], LH, lane);
        float8 g0[8];
#pragma unroll
        for (int t8 = 0; t8 < 8; ++t8) {
            half16 bf = load_fragB(&sWhh0[t8 * 16 * LH], LH, lane);
            float8 c;
#pragma unroll
            for (int r = 0; r < 8; ++r) c[r] = bs0[t8] + w0c[t8] * x0[r] + w1c[t8] * x1[r];
            g0[t8] = wmma_f16(a0, bf, c);
        }
        // cell update, store h0 (f16) to per-wave LDS for re-fragmenting
#pragma unroll
        for (int jh = 0; jh < 2; ++jh) {
            float8 gi = g0[jh], gf = g0[2 + jh], gg = g0[4 + jh], go = g0[6 + jh];
#pragma unroll
            for (int r = 0; r < 8; ++r) {
                float cc = fast_sig(gf[r]) * c0[jh][r] + fast_sig(gi[r]) * fast_tanh(gg[r]);
                c0[jh][r] = cc;
                float hv = fast_sig(go[r]) * fast_tanh(cc);
                sH0[wave][(moff + r) * LH + jh * 16 + (lane & 15)] = (_Float16)hv;
            }
        }

        // ---- layer 1: gates = bias + h0new@Wih1^T + h1@Whh1^T (two K=32 WMMAs)
        half16 ain = load_fragA(&sH0[wave][0], LH, lane);
        half16 ah1 = load_fragA(&sH1[wave][0], LH, lane);
        float8 g1[8];
#pragma unroll
        for (int t8 = 0; t8 < 8; ++t8) {
            half16 bfi = load_fragB(&sWih1[t8 * 16 * LH], LH, lane);
            half16 bfh = load_fragB(&sWhh1[t8 * 16 * LH], LH, lane);
            float8 c;
#pragma unroll
            for (int r = 0; r < 8; ++r) c[r] = bs1[t8];
            c = wmma_f16(ain, bfi, c);
            c = wmma_f16(ah1, bfh, c);
            g1[t8] = c;
        }
#pragma unroll
        for (int jh = 0; jh < 2; ++jh) {
            float8 gi = g1[jh], gf = g1[2 + jh], gg = g1[4 + jh], go = g1[6 + jh];
#pragma unroll
            for (int r = 0; r < 8; ++r) {
                float cc = fast_sig(gf[r]) * c1[jh][r] + fast_sig(gi[r]) * fast_tanh(gg[r]);
                c1[jh][r] = cc;
                float hv = fast_sig(go[r]) * fast_tanh(cc);
                sH1[wave][(moff + r) * LH + jh * 16 + (lane & 15)] = (_Float16)hv;
            }
        }
    }

    __syncthreads();
    // out_ = h1_final @ Wl^T + bl  (32->2, VALU)
    if (tid < 128) {
        const _Float16* hrow = &sH1[tid >> 4][(tid & 15) * LH];
        float a0 = bl[0], a1 = bl[1];
#pragma unroll
        for (int k = 0; k < LH; ++k) {
            float hv = (float)hrow[k];
            a0 += hv * Wl[k];
            a1 += hv * Wl[LH + k];
        }
        outL[(size_t)(gbase + tid) * 2 + 0] = a0;
        outL[(size_t)(gbase + tid) * 2 + 1] = a1;
    }
}

// ---------------------------------------------------------------------------
// Kernel 0: convert MLP weights f32 -> f16 into workspace (W1 K-padded to 32)
// ---------------------------------------------------------------------------
__global__ __launch_bounds__(256)
void convert_kernel(const float* __restrict__ W1, const float* __restrict__ W2,
                    const float* __restrict__ W3, const float* __restrict__ W4,
                    _Float16* __restrict__ W1h, _Float16* __restrict__ W2h,
                    _Float16* __restrict__ W3h, _Float16* __restrict__ W4h)
{
    int i = blockIdx.x * 256 + threadIdx.x;   // 65536 threads
    if (i < HDIM * 32) {
        int r = i >> 5, c = i & 31;
        W1h[i] = (_Float16)((c < 30) ? W1[r * 30 + c] : 0.0f);
    }
    if (i < HDIM * HDIM) { W2h[i] = (_Float16)W2[i]; W3h[i] = (_Float16)W3[i]; }
    if (i < HDIM) W4h[i] = (_Float16)W4[i];
}

// ---------------------------------------------------------------------------
// Kernel 2: MLP head 30->256->256->256->1, all WMMA. One wave = 16 rows.
// ---------------------------------------------------------------------------
__global__ __launch_bounds__(64)
void mlp_kernel(const float* __restrict__ state, const float* __restrict__ action,
                const float* __restrict__ outL,
                const _Float16* __restrict__ W1h, const float* __restrict__ b1,
                const _Float16* __restrict__ W2h, const float* __restrict__ b2,
                const _Float16* __restrict__ W3h, const float* __restrict__ b3,
                const _Float16* __restrict__ W4h, const float* __restrict__ b4,
                float* __restrict__ out)
{
    __shared__ _Float16 sXin[2][16 * 32];
    __shared__ _Float16 sA[2][2][16 * HDIM];   // per-wave ping-pong activation tiles

    const int tid   = threadIdx.x;
    const int wave  = tid >> 5;
    const int lane  = tid & 31;
    const int mbase = blockIdx.x * 32 + wave * 16;
    const int moff  = (lane < 16) ? 0 : 8;

    // Stage input row [state(cols 8,9 <- outL), action, pad] as f16
    {
        int m  = lane & 15;
        int gm = mbase + m;
        int ch = (lane >> 4) * 16;
#pragma unroll
        for (int c = 0; c < 16; ++c) {
            int cc = ch + c;
            float v;
            if (cc < NS) {
                if (cc == 8)      v = outL[(size_t)gm * 2 + 0];
                else if (cc == 9) v = outL[(size_t)gm * 2 + 1];
                else              v = state[(size_t)gm * NS + cc];
            } else if (cc < NS + NA) v = action[(size_t)gm * NA + (cc - NS)];
            else v = 0.0f;
            sXin[wave][m * 32 + cc] = (_Float16)v;
        }
    }

    _Float16* act0 = &sA[wave][0][0];
    _Float16* act1 = &sA[wave][1][0];

    // ---- layer 1: (K=32 padded) single WMMA per 16-col tile
    half16 ax = load_fragA(&sXin[wave][0], 32, lane);
    for (int t16 = 0; t16 < 16; ++t16) {
        float  bn = b1[t16 * 16 + (lane & 15)];
        half16 bf = load_fragB(W1h + t16 * 16 * 32, 32, lane);
        float8 c;
#pragma unroll
        for (int r = 0; r < 8; ++r) c[r] = bn;
        c = wmma_f16(ax, bf, c);
#pragma unroll
        for (int r = 0; r < 8; ++r)
            act0[(moff + r) * HDIM + t16 * 16 + (lane & 15)] = (_Float16)fmaxf(c[r], 0.0f);
    }

    // ---- layer 2: 256->256, 8 K-chunks; B-frags stream from global (L2-resident)
    {
        half16 aL[8];
#pragma unroll
        for (int k = 0; k < 8; ++k) aL[k] = load_fragA(act0 + k * 32, HDIM, lane);
        for (int t16 = 0; t16 < 16; ++t16) {
            float  bn = b2[t16 * 16 + (lane & 15)];
            float8 c;
#pragma unroll
            for (int r = 0; r < 8; ++r) c[r] = bn;
#pragma unroll
            for (int k = 0; k < 8; ++k) {
                half16 bf = load_fragB(W2h + t16 * 16 * HDIM + k * 32, HDIM, lane);
                c = wmma_f16(aL[k], bf, c);
            }
#pragma unroll
            for (int r = 0; r < 8; ++r)
                act1[(moff + r) * HDIM + t16 * 16 + (lane & 15)] = (_Float16)fmaxf(c[r], 0.0f);
        }
    }

    // ---- layer 3: 256->256
    {
        half16 aL[8];
#pragma unroll
        for (int k = 0; k < 8; ++k) aL[k] = load_fragA(act1 + k * 32, HDIM, lane);
        for (int t16 = 0; t16 < 16; ++t16) {
            float  bn = b3[t16 * 16 + (lane & 15)];
            float8 c;
#pragma unroll
            for (int r = 0; r < 8; ++r) c[r] = bn;
#pragma unroll
            for (int k = 0; k < 8; ++k) {
                half16 bf = load_fragB(W3h + t16 * 16 * HDIM + k * 32, HDIM, lane);
                c = wmma_f16(aL[k], bf, c);
            }
#pragma unroll
            for (int r = 0; r < 8; ++r)
                act0[(moff + r) * HDIM + t16 * 16 + (lane & 15)] = (_Float16)fmaxf(c[r], 0.0f);
        }
    }

    // ---- layer 4: 256->1 as WMMA with B nonzero only in column n=0
    {
        half16 aL[8];
#pragma unroll
        for (int k = 0; k < 8; ++k) aL[k] = load_fragA(act0 + k * 32, HDIM, lane);
        float  b4v = b4[0];
        half16 z   = (half16){};
        float8 c;
#pragma unroll
        for (int r = 0; r < 8; ++r) c[r] = ((lane & 15) == 0) ? b4v : 0.0f;
#pragma unroll
        for (int k = 0; k < 8; ++k) {
            const _Float16* wp = W4h + k * 32 + ((lane < 16) ? 0 : 16);
            half8 lo = *(const half8*)(wp);
            half8 hi = *(const half8*)(wp + 8);
            half16 wf = __builtin_shufflevector(lo, hi, 0,1,2,3,4,5,6,7,8,9,10,11,12,13,14,15);
            half16 bf = ((lane & 15) == 0) ? wf : z;
            c = wmma_f16(aL[k], bf, c);
        }
        if ((lane & 15) == 0) {
#pragma unroll
            for (int r = 0; r < 8; ++r)
                out[mbase + moff + r] = c[r];
        }
    }
}

// ---------------------------------------------------------------------------
extern "C" void kernel_launch(void* const* d_in, const int* in_sizes, int n_in,
                              void* d_out, int out_size, void* d_ws, size_t ws_size,
                              hipStream_t stream)
{
    const float* state  = (const float*)d_in[0];
    const float* action = (const float*)d_in[1];
    const float* xseq   = (const float*)d_in[2];
    const float* Wih0   = (const float*)d_in[3];
    const float* Whh0   = (const float*)d_in[4];
    const float* bih0   = (const float*)d_in[5];
    const float* bhh0   = (const float*)d_in[6];
    const float* Wih1   = (const float*)d_in[7];
    const float* Whh1   = (const float*)d_in[8];
    const float* bih1   = (const float*)d_in[9];
    const float* bhh1   = (const float*)d_in[10];
    const float* Wl     = (const float*)d_in[11];
    const float* bl     = (const float*)d_in[12];
    const float* W1     = (const float*)d_in[13];
    const float* b1     = (const float*)d_in[14];
    const float* W2     = (const float*)d_in[15];
    const float* b2     = (const float*)d_in[16];
    const float* W3     = (const float*)d_in[17];
    const float* b3     = (const float*)d_in[18];
    const float* W4     = (const float*)d_in[19];
    const float* b4     = (const float*)d_in[20];

    char* ws = (char*)d_ws;
    float*     outL = (float*)ws;                      // [B,2] f32 (512 KB)
    _Float16*  W1h  = (_Float16*)(ws + 524288);        // 256x32  f16
    _Float16*  W2h  = (_Float16*)(ws + 540672);        // 256x256 f16
    _Float16*  W3h  = (_Float16*)(ws + 671744);        // 256x256 f16
    _Float16*  W4h  = (_Float16*)(ws + 802816);        // 256     f16

    convert_kernel<<<256, 256, 0, stream>>>(W1, W2, W3, W4, W1h, W2h, W3h, W4h);
    lstm_kernel<<<B_TOT / 128, 256, 0, stream>>>(xseq, Wih0, Whh0, bih0, bhh0,
                                                 Wih1, Whh1, bih1, bhh1, Wl, bl, outL);
    mlp_kernel<<<B_TOT / 32, 64, 0, stream>>>(state, action, outL,
                                              W1h, b1, W2h, b2, W3h, b3, W4h, b4,
                                              (float*)d_out);
}
